// StackedLSTM_6665789243438
// MI455X (gfx1250) — matrix operand order repaired and checked by reading, not compiled, not run on hardware
//
#include <hip/hip_runtime.h>
#include <hip/hip_bf16.h>

typedef __bf16 bf16;
typedef __attribute__((ext_vector_type(16))) __bf16 v16bf;
typedef __attribute__((ext_vector_type(8)))  __bf16 v8bf;
typedef __attribute__((ext_vector_type(8)))  float  v8f;

#define B_  64
#define T_  256
#define H_  512
#define L_  2

// ---------------------------------------------------------------------------
// WMMA fragment loaders (layouts per CDNA5 ISA 7.12.2)
// ---------------------------------------------------------------------------

// A-matrix 16x32 bf16: lane l -> row M=l&15, hi=l>>4.
// element e<8  -> K = kb + hi*8 + e
// element e>=8 -> K = kb + 16 + hi*8 + (e-8)
__device__ __forceinline__ v16bf load_a_frag(const bf16* __restrict__ row, int kb, int hi) {
  v8bf lo = *(const v8bf*)(row + kb + hi * 8);
  v8bf hs = *(const v8bf*)(row + kb + 16 + hi * 8);
  v16bf r;
#pragma unroll
  for (int i = 0; i < 8; ++i) { r[i] = lo[i]; r[i + 8] = hs[i]; }
  return r;
}

// B-matrix 32x16 bf16: lane l -> col N=l&15, hi=l>>4.
// element e<8 -> K = kb + hi*8 + e, e>=8 -> K = kb + 16 + hi*8 + (e-8)   (row ptr points at W[gate][N][*], contiguous in K)
__device__ __forceinline__ v16bf load_b_frag(const bf16* __restrict__ row, int kb, int hi) {
  v8bf lo = *(const v8bf*)(row + kb + hi * 8);
  v8bf hs = *(const v8bf*)(row + kb + 16 + hi * 8);
  v16bf r;
#pragma unroll
  for (int i = 0; i < 8; ++i) { r[i] = lo[i]; r[i + 8] = hs[i]; }
  return r;
}

__device__ __forceinline__ float sigf(float x) { return 1.0f / (1.0f + __expf(-x)); }

// ---------------------------------------------------------------------------
// Setup kernels (run every call; workspace is not persistent)
// ---------------------------------------------------------------------------

__global__ void cvt_weights_kernel(const float* __restrict__ wx, const float* __restrict__ wh,
                                   bf16* __restrict__ wxb, bf16* __restrict__ whb, int n) {
  int i = blockIdx.x * blockDim.x + threadIdx.x;
  if (i >= n) return;
  wxb[i] = (bf16)wx[i];
  whb[i] = (bf16)wh[i];
}

__global__ void bias_sum_kernel(const float* __restrict__ bx, const float* __restrict__ bh,
                                float* __restrict__ bsum, int n) {
  int i = blockIdx.x * blockDim.x + threadIdx.x;
  if (i >= n) return;
  bsum[i] = bx[i] + bh[i];
}

__global__ void embed_cvt_kernel(const int* __restrict__ idx, const float* __restrict__ emb,
                                 bf16* __restrict__ X0, int n) {
  int i = blockIdx.x * blockDim.x + threadIdx.x;
  if (i >= n) return;
  int row = i / H_;
  int k   = i - row * H_;
  X0[i] = (bf16)emb[(size_t)idx[row] * H_ + k];
}

__global__ void init_state_kernel(bf16* __restrict__ h0, float* __restrict__ c0,
                                  float* __restrict__ pooled, int n) {
  int i = blockIdx.x * blockDim.x + threadIdx.x;
  if (i >= n) return;
  h0[i] = (bf16)0.0f;
  c0[i] = 0.0f;
  pooled[i] = 0.0f;
}

// ---------------------------------------------------------------------------
// One LSTM time-step. 32 blocks x 4 waves; each wave owns one
// (batch-tile bt, col-tile jt) and computes ALL FOUR gate tiles over
// K = 1024 (x_t part then h part), reusing the A fragment across gates.
//
// Software pipeline with prefetch distance 2 (3 register fragment slots),
// pinned with __builtin_amdgcn_sched_barrier(0) between the load group and
// the WMMA group of each unrolled iteration so the backend cannot sink the
// prefetch loads down to their consumers. This keeps ~2 K-steps of loads in
// flight -> partial s_wait_loadcnt instead of full drains before each WMMA.
// __launch_bounds__(128,1): occupancy is launch-limited (128 waves/step),
// so the ~170-VGPR pipeline is free.
//
// Gate pointwise is pure register VALU work (i/f/g/o of one (M,N) share a
// lane & acc index): no LDS, no barrier, no divergence before WMMA.
// ---------------------------------------------------------------------------
__global__ __launch_bounds__(128, 1)
void lstm_step_kernel(const bf16*  __restrict__ Xin,    // [B,T,H] layer input (bf16)
                      const bf16*  __restrict__ hin,    // [B,H]
                      const float* __restrict__ cin,    // [B,H]
                      const bf16*  __restrict__ wxb,    // [4,H,H] layer weights (bf16)
                      const bf16*  __restrict__ whb,    // [4,H,H]
                      const float* __restrict__ bsum,   // [4,H] bx+bh
                      bf16*        __restrict__ hout,   // [B,H]  (gets c_new: swap bug)
                      float*       __restrict__ cout,   // [B,H]  (gets h_new: swap bug)
                      bf16*        __restrict__ Xout,   // [B,T,H] o-gate out (layer 0) or null
                      float*       __restrict__ pooled, // [B,H] o-gate accum (layer 1) or null
                      int t)
{
  const int lane = threadIdx.x & 31;
  const int wave = threadIdx.x >> 5;
  const int w    = blockIdx.x * 4 + wave;  // 0..127 global wave id
  const int jt   = w & 31;                 // col tile (16 cols of H=512)
  const int bt   = w >> 5;                 // batch tile (16 rows of B=64)
  const int mn   = lane & 15;
  const int hi   = lane >> 4;

  const bf16* xrow = Xin + ((size_t)(bt * 16 + mn) * T_ + t) * H_;
  const bf16* hrow = hin + (size_t)(bt * 16 + mn) * H_;
  // B operand row = W[gate][N][*]; N = jt*16 + mn, contiguous in K.
  const size_t wrow = (size_t)(jt * 16 + mn) * H_;
  const bf16* wxr = wxb + wrow;   // gate g at +g*H*H
  const bf16* whr = whb + wrow;

  v8f acc0 = {0.f,0.f,0.f,0.f,0.f,0.f,0.f,0.f};
  v8f acc1 = acc0, acc2 = acc0, acc3 = acc0;

  const size_t GS = (size_t)H_ * H_;   // gate stride in elements

  // 32 K-steps total: s<16 -> x part (Wx), s>=16 -> h part (Wh).
  v16bf af[3];        // A fragment slots
  v16bf bfr[3][4];    // B fragment slots (4 gates)

#define LOAD_STEP(sn, slot)                                                    \
  do {                                                                         \
    const bf16* ar_ = ((sn) < 16) ? xrow : hrow;                               \
    const bf16* br_ = ((sn) < 16) ? wxr  : whr;                                \
    const int   kb_ = ((sn) & 15) * 32;                                        \
    af[slot]     = load_a_frag(ar_, kb_, hi);                                  \
    bfr[slot][0] = load_b_frag(br_ + 0 * GS, kb_, hi);                         \
    bfr[slot][1] = load_b_frag(br_ + 1 * GS, kb_, hi);                         \
    bfr[slot][2] = load_b_frag(br_ + 2 * GS, kb_, hi);                         \
    bfr[slot][3] = load_b_frag(br_ + 3 * GS, kb_, hi);                         \
  } while (0)

  // Prologue: fill slots for steps 0 and 1.
  LOAD_STEP(0, 0);
  LOAD_STEP(1, 1);

#pragma unroll
  for (int s = 0; s < 32; ++s) {
    const int cs = s % 3;         // current slot
    const int ps = (s + 2) % 3;   // prefetch slot (step s+2)
    if (s + 2 < 32) LOAD_STEP(s + 2, ps);
    __builtin_amdgcn_sched_barrier(0);   // keep prefetch loads above the WMMAs
    acc0 = __builtin_amdgcn_wmma_f32_16x16x32_bf16(false, af[cs], false, bfr[cs][0],
                                                   (short)0, acc0, false, false);
    acc1 = __builtin_amdgcn_wmma_f32_16x16x32_bf16(false, af[cs], false, bfr[cs][1],
                                                   (short)0, acc1, false, false);
    acc2 = __builtin_amdgcn_wmma_f32_16x16x32_bf16(false, af[cs], false, bfr[cs][2],
                                                   (short)0, acc2, false, false);
    acc3 = __builtin_amdgcn_wmma_f32_16x16x32_bf16(false, af[cs], false, bfr[cs][3],
                                                   (short)0, acc3, false, false);
    __builtin_amdgcn_sched_barrier(0);   // keep WMMAs from drifting past next loads
  }
#undef LOAD_STEP

  // Pointwise: lane -> N = mn, acc index r -> M = r + hi*8.
  // All four gates for (M,N) sit in this lane at index r: pure register math.
  const int col = jt * 16 + mn;
  const float bi = bsum[0 * H_ + col];
  const float bf = bsum[1 * H_ + col];
  const float bg = bsum[2 * H_ + col];
  const float bo = bsum[3 * H_ + col];

#pragma unroll
  for (int r = 0; r < 8; ++r) {
    const int b = bt * 16 + r + hi * 8;
    const size_t sidx = (size_t)b * H_ + col;

    float i_ = sigf(acc0[r] + bi);
    float f_ = sigf(acc1[r] + bf);
    float g_ = tanhf(acc2[r] + bg);
    float o_ = sigf(acc3[r] + bo);

    float c  = cin[sidx];
    float cn = f_ * c + i_ * g_;
    float hn = o_ * tanhf(cn);

    hout[sidx] = (bf16)cn;  // reference bug: next h <- c_new (feeds next GEMM as bf16)
    cout[sidx] = hn;        // reference bug: next c <- h_new
    if (Xout)   Xout[((size_t)b * T_ + t) * H_ + col] = (bf16)o_;
    if (pooled) pooled[sidx] += o_;   // unique owner per (b,col) per step: no atomics
  }
}

// ---------------------------------------------------------------------------
// out[b] = sigmoid( mean_T(o)[b,:] . wo + bo )
// ---------------------------------------------------------------------------
__global__ void final_kernel(const float* __restrict__ pooled, const float* __restrict__ wo,
                             const float* __restrict__ bo, float* __restrict__ out) {
  int b = blockIdx.x * blockDim.x + threadIdx.x;
  if (b >= B_) return;
  float s = 0.0f;
  for (int j = 0; j < H_; ++j) s += pooled[(size_t)b * H_ + j] * wo[j];
  out[b] = sigf(s * (1.0f / (float)T_) + bo[0]);
}

// ---------------------------------------------------------------------------
extern "C" void kernel_launch(void* const* d_in, const int* in_sizes, int n_in,
                              void* d_out, int out_size, void* d_ws, size_t ws_size,
                              hipStream_t stream) {
  (void)in_sizes; (void)n_in; (void)out_size; (void)ws_size;
  const int*   idx = (const int*)  d_in[0];
  const float* emb = (const float*)d_in[1];
  const float* wx  = (const float*)d_in[2];
  const float* bx  = (const float*)d_in[3];
  const float* wh  = (const float*)d_in[4];
  const float* bh  = (const float*)d_in[5];
  const float* wo  = (const float*)d_in[6];
  const float* bo  = (const float*)d_in[7];
  float* out = (float*)d_out;

  char* ws = (char*)d_ws;
  size_t off = 0;
  auto carve = [&](size_t bytes) -> char* {
    char* p = ws + off;
    off += (bytes + 255) & ~(size_t)255;
    return p;
  };

  bf16*  wxb    = (bf16*) carve((size_t)L_ * 4 * H_ * H_ * sizeof(bf16));   // 4 MiB
  bf16*  whb    = (bf16*) carve((size_t)L_ * 4 * H_ * H_ * sizeof(bf16));   // 4 MiB
  float* bsum   = (float*)carve((size_t)L_ * 4 * H_ * sizeof(float));
  bf16*  X0     = (bf16*) carve((size_t)B_ * T_ * H_ * sizeof(bf16));       // 16 MiB
  bf16*  X1     = (bf16*) carve((size_t)B_ * T_ * H_ * sizeof(bf16));       // 16 MiB
  bf16*  hb0    = (bf16*) carve((size_t)B_ * H_ * sizeof(bf16));
  bf16*  hb1    = (bf16*) carve((size_t)B_ * H_ * sizeof(bf16));
  float* cb0    = (float*)carve((size_t)B_ * H_ * sizeof(float));
  float* cb1    = (float*)carve((size_t)B_ * H_ * sizeof(float));
  float* pooled = (float*)carve((size_t)B_ * H_ * sizeof(float));
  bf16*  hb[2]  = {hb0, hb1};
  float* cb[2]  = {cb0, cb1};

  { int N = L_ * 4 * H_ * H_;
    cvt_weights_kernel<<<(N + 255) / 256, 256, 0, stream>>>(wx, wh, wxb, whb, N); }
  { int N = L_ * 4 * H_;
    bias_sum_kernel<<<(N + 255) / 256, 256, 0, stream>>>(bx, bh, bsum, N); }
  { int N = B_ * T_ * H_;
    embed_cvt_kernel<<<(N + 255) / 256, 256, 0, stream>>>(idx, emb, X0, N); }
  { int N = B_ * H_;
    init_state_kernel<<<(N + 255) / 256, 256, 0, stream>>>(hb[0], cb[0], pooled, N); }

  int s = 0;
  for (int l = 0; l < L_; ++l) {
    const bf16*  Xin  = (l == 0) ? X0 : X1;
    bf16*        Xout = (l == 0) ? X1 : nullptr;
    float*       pool = (l == 0) ? nullptr : pooled;
    const bf16*  wxl  = wxb + (size_t)l * 4 * H_ * H_;
    const bf16*  whl  = whb + (size_t)l * 4 * H_ * H_;
    const float* bl   = bsum + (size_t)l * 4 * H_;
    for (int t = 0; t < T_; ++t, ++s) {
      lstm_step_kernel<<<32, 128, 0, stream>>>(Xin, hb[s & 1], cb[s & 1],
                                               wxl, whl, bl,
                                               hb[(s + 1) & 1], cb[(s + 1) & 1],
                                               Xout, pool, t);
    }
  }

  final_kernel<<<1, 64, 0, stream>>>(pooled, wo, bo, out);
}
